// Net_15693810499811
// MI455X (gfx1250) — matrix operand order repaired and checked by reading, not compiled, not run on hardware
//
#include <hip/hip_runtime.h>
#include <hip/hip_bf16.h>

// CDNA5 (gfx1250) point-transformer GNN pipeline.
// All dense GEMMs (node lin/lin_src/lin_dst, neck, attn_nn per-edge, head1)
// run on v_wmma_f32_16x16x32_f16 with f16 operands / f32 accumulation.
// Register-blocked: GEMM waves keep 4 M-subtile accumulators (B reused 4x),
// edge-conv waves keep 8 N-tile accumulators (A fragment reused 8x).

typedef _Float16 v16h __attribute__((ext_vector_type(16)));
typedef _Float16 v8h  __attribute__((ext_vector_type(8)));
typedef float    v8f  __attribute__((ext_vector_type(8)));

#define BN_EPS 1e-5f

// ---------------------------------------------------------------------------
// Pack W [K,N] f32 (row-major) into WMMA B-fragment order f16:
//   frag[kt][nt][lane][e]  where lane L: K = kt*32 + (L/16)*16 + e, N = nt*16 + L%16
// ---------------------------------------------------------------------------
__global__ void pack_b_kernel(const float* __restrict__ W, _Float16* __restrict__ out,
                              int K, int Kpad, int N) {
  int idx = blockIdx.x * blockDim.x + threadIdx.x;
  int tot = Kpad * N;
  if (idx >= tot) return;
  int k = idx / N, n = idx % N;
  int kt = k >> 5, kr = k & 31;
  int lane = ((kr >> 4) << 4) | (n & 15);
  int e    = kr & 15;
  int nt   = n >> 4;
  float v = (k < K) ? W[(size_t)k * N + n] : 0.0f;
  out[(((size_t)(kt * (N >> 4) + nt) * 32 + lane) << 4) + e] = (_Float16)v;
}

// ---------------------------------------------------------------------------
// Fuse Linear-bias + BatchNorm(eval) into per-channel scale/shift:
//   y = relu((x@W + b - rm) * g/sqrt(rv+eps) + be)  =  relu((x@W)*sc + sh)
// ---------------------------------------------------------------------------
__global__ void fuse_bn_kernel(const float* __restrict__ b,  const float* __restrict__ be,
                               const float* __restrict__ g,  const float* __restrict__ rm,
                               const float* __restrict__ rv, float* __restrict__ ss, int d) {
  int i = blockIdx.x * blockDim.x + threadIdx.x;
  if (i >= d) return;
  float sc = g[i] * rsqrtf(rv[i] + BN_EPS);
  ss[i]     = sc;
  ss[d + i] = be[i] + (b[i] - rm[i]) * sc;
}

// ---------------------------------------------------------------------------
// CSR ranges for dst-sorted edge lists (self loops guarantee >=1 edge/node).
// ---------------------------------------------------------------------------
__global__ void edge_range_init(int* __restrict__ rs, int* __restrict__ re, int n) {
  int i = blockIdx.x * blockDim.x + threadIdx.x;
  if (i < n) { rs[i] = 0; re[i] = 0; }
}
__global__ void edge_range_build(const int* __restrict__ dst, int E,
                                 int* __restrict__ rs, int* __restrict__ re) {
  int i = blockIdx.x * blockDim.x + threadIdx.x;
  if (i >= E) return;
  int d = dst[i];
  if (i == 0     || dst[i - 1] != d) rs[d] = i;
  if (i == E - 1 || dst[i + 1] != d) re[d] = i + 1;
}

// ---------------------------------------------------------------------------
// Generic WMMA GEMM: C[m, col0+n] = epilogue(A[M,K] @ Bpacked[Kpad,N])
// Block = 128 threads = 4 waves; block covers 64 rows x 64 cols.
// Each wave owns one 16-col tile and register-blocks 4 M-subtiles, reusing
// every B fragment for 4 WMMAs. Grid (M/64, N/64).
// A staged in LDS as f16 with zero K-padding; ss != nullptr -> relu(v*sc+sh).
// ---------------------------------------------------------------------------
__global__ void wmma_gemm_kernel(const float* __restrict__ A, int K, int Kpad,
                                 const _Float16* __restrict__ Bp, int N,
                                 const float* __restrict__ ss,
                                 float* __restrict__ Co, int ldo, int col0) {
  extern __shared__ _Float16 shA[];            // [64][Kpad]
  const int tid  = threadIdx.x;
  const int m0   = blockIdx.x * 64;
  const int lane = tid & 31, wv = tid >> 5;
  const int lm   = lane & 15, gh = lane >> 4;

  for (int idx = tid; idx < 64 * Kpad; idx += 128) {
    int m = idx / Kpad, k = idx - m * Kpad;
    shA[idx] = (k < K) ? (_Float16)A[(size_t)(m0 + m) * K + k] : (_Float16)0.0f;
  }
  __syncthreads();

  const int nt = blockIdx.y * 4 + wv;          // 16-column tile index
  const int ntiles = N >> 4;
  v8f acc[4] = {};
  for (int kt = 0; kt < (Kpad >> 5); ++kt) {
    v16h b = *(const v16h*)(Bp + ((size_t)(kt * ntiles + nt) * 32 + lane) * 16);
#pragma unroll
    for (int mt = 0; mt < 4; ++mt) {
      v8h lo = *(const v8h*)&shA[(mt * 16 + lm) * Kpad + kt * 32 + gh * 16];
      v8h hi = *(const v8h*)&shA[(mt * 16 + lm) * Kpad + kt * 32 + gh * 16 + 8];
      v16h a;
#pragma unroll
      for (int i = 0; i < 8; ++i) { a[i] = lo[i]; a[8 + i] = hi[i]; }
      acc[mt] = __builtin_amdgcn_wmma_f32_16x16x32_f16(false, a, false, b, (short)0,
                                                       acc[mt], false, false);
    }
  }
  const int n = nt * 16 + lm;
  float sc = 1.0f, sh = 0.0f;
  if (ss) { sc = ss[n]; sh = ss[N + n]; }
#pragma unroll
  for (int mt = 0; mt < 4; ++mt) {
#pragma unroll
    for (int i = 0; i < 8; ++i) {
      int row = m0 + mt * 16 + gh * 8 + i;
      float v = acc[mt][i];
      if (ss) v = fmaxf(v * sc + sh, 0.0f);
      Co[(size_t)row * ldo + col0 + n] = v;
    }
  }
}

// ---------------------------------------------------------------------------
// PointTransformerConv per destination node. One workgroup (128 thr, 4 waves)
// per dst. Builds t = a_d[d]-a_s[s]+delta in LDS (f16), runs attn_nn CxC GEMM
// with WMMA (kt-outer, 8 N-tile accumulators so each A fragment feeds 8 WMMAs),
// then per-channel scatter-softmax + weighted (xv[src]+delta) aggregation.
// ---------------------------------------------------------------------------
template <int C>
__global__ void edge_conv_kernel(const float* __restrict__ a_s, const float* __restrict__ a_d,
                                 const float* __restrict__ xv,
                                 const float* __restrict__ pos, const float* __restrict__ nrm,
                                 const int* __restrict__ esrc,
                                 const int* __restrict__ rs, const int* __restrict__ re,
                                 const float* __restrict__ posW, const float* __restrict__ posSS,
                                 const _Float16* __restrict__ attnWp,
                                 const float* __restrict__ attnSS,
                                 float* __restrict__ out, int ldo, int col0) {
  __shared__ _Float16 shT[64 * C];     // gathered GEMM input (f16)
  __shared__ float    shAl[64 * 128];  // alpha for one 128-channel block
  __shared__ float    shAd[C];         // a_d row of this dst
  __shared__ float    shRel[64 * 8];   // rel-pos/rel-normal per edge
  __shared__ int      shSrc[64];

  const int d   = blockIdx.x;
  const int tid = threadIdx.x;
  int start = rs[d], end = re[d];
  int cnt = end - start;
  if (cnt > 64) cnt = 64;
  if (cnt <= 0) {
    for (int c = tid; c < C; c += 128) out[(size_t)d * ldo + col0 + c] = 0.0f;
    return;
  }
  const int cntPad = (cnt + 15) & ~15;

  for (int k = tid; k < C; k += 128) shAd[k] = a_d[(size_t)d * C + k];
  if (tid < cnt) {
    int s = esrc[start + tid];
    shSrc[tid] = s;
    shRel[tid * 8 + 0] = pos[d * 3 + 0] - pos[s * 3 + 0];
    shRel[tid * 8 + 1] = pos[d * 3 + 1] - pos[s * 3 + 1];
    shRel[tid * 8 + 2] = pos[d * 3 + 2] - pos[s * 3 + 2];
    shRel[tid * 8 + 3] = nrm[d * 3 + 0] - nrm[s * 3 + 0];
    shRel[tid * 8 + 4] = nrm[d * 3 + 1] - nrm[s * 3 + 1];
    shRel[tid * 8 + 5] = nrm[d * 3 + 2] - nrm[s * 3 + 2];
  }
  __syncthreads();

  // Build t rows (zero-padded to cntPad).
  for (int idx = tid; idx < cntPad * C; idx += 128) {
    int j = idx / C, k = idx - j * C;
    float v = 0.0f;
    if (j < cnt) {
      int s = shSrc[j];
      float dot = 0.0f;
#pragma unroll
      for (int i = 0; i < 6; ++i) dot += shRel[j * 8 + i] * posW[i * C + k];
      float delta = fmaxf(dot * posSS[k] + posSS[C + k], 0.0f);
      v = shAd[k] - a_s[(size_t)s * C + k] + delta;
    }
    shT[j * C + k] = (_Float16)v;
  }
  __syncthreads();

  const int lane = tid & 31, wv = tid >> 5;
  const int lm   = lane & 15, gh = lane >> 4;
  const int jtiles = cntPad >> 4;

  for (int nb = 0; nb < C / 128; ++nb) {
    if (wv < jtiles) {
      const int arow = wv * 16 + lm;           // edge row for this lane's A frag
      v8f acc[8] = {};
      for (int kt = 0; kt < (C >> 5); ++kt) {
        v8h lo = *(const v8h*)&shT[arow * C + kt * 32 + gh * 16];
        v8h hi = *(const v8h*)&shT[arow * C + kt * 32 + gh * 16 + 8];
        v16h a;
#pragma unroll
        for (int i = 0; i < 8; ++i) { a[i] = lo[i]; a[8 + i] = hi[i]; }
#pragma unroll
        for (int ntl = 0; ntl < 8; ++ntl) {
          v16h b = *(const v16h*)(attnWp +
                                  ((size_t)(kt * (C >> 4) + nb * 8 + ntl) * 32 + lane) * 16);
          acc[ntl] = __builtin_amdgcn_wmma_f32_16x16x32_f16(false, a, false, b, (short)0,
                                                            acc[ntl], false, false);
        }
      }
#pragma unroll
      for (int ntl = 0; ntl < 8; ++ntl) {
        const int n = nb * 128 + ntl * 16 + lm;
        const float sc = attnSS[n], sh = attnSS[C + n];
#pragma unroll
        for (int i = 0; i < 8; ++i) {
          int jr = wv * 16 + gh * 8 + i;
          shAl[jr * 128 + ntl * 16 + lm] = fmaxf(acc[ntl][i] * sc + sh, 0.0f);
        }
      }
    }
    __syncthreads();

    // Scatter-softmax over edges + aggregation: one thread per channel.
    {
      const int gc = nb * 128 + tid;
      const float w0 = posW[0 * C + gc], w1 = posW[1 * C + gc], w2 = posW[2 * C + gc];
      const float w3 = posW[3 * C + gc], w4 = posW[4 * C + gc], w5 = posW[5 * C + gc];
      const float psc = posSS[gc], psh = posSS[C + gc];
      float m = -3.402823e38f;
      for (int j = 0; j < cnt; ++j) m = fmaxf(m, shAl[j * 128 + tid]);
      float den = 0.0f, num = 0.0f;
      for (int j = 0; j < cnt; ++j) {
        float e = __expf(shAl[j * 128 + tid] - m);
        int s = shSrc[j];
        float dot = shRel[j * 8 + 0] * w0 + shRel[j * 8 + 1] * w1 + shRel[j * 8 + 2] * w2 +
                    shRel[j * 8 + 3] * w3 + shRel[j * 8 + 4] * w4 + shRel[j * 8 + 5] * w5;
        float delta = fmaxf(dot * psc + psh, 0.0f);
        num += e * (xv[(size_t)s * C + gc] + delta);
        den += e;
      }
      out[(size_t)d * ldo + col0 + gc] = num / den;
    }
    __syncthreads();
  }
}

// ---------------------------------------------------------------------------
// Residue pooling: max over APR atoms per residue (512 ch), mean pos, mask.
// ---------------------------------------------------------------------------
__global__ void pool_kernel(const float* __restrict__ nout, const float* __restrict__ pos,
                            const int* __restrict__ mask, int apr, int Cn,
                            float* __restrict__ resf, float* __restrict__ aapos,
                            int* __restrict__ maskr) {
  const int r = blockIdx.x, tid = threadIdx.x;
  for (int c = tid; c < Cn; c += blockDim.x) {
    float m = -3.402823e38f;
    for (int a = 0; a < apr; ++a)
      m = fmaxf(m, nout[(size_t)(r * apr + a) * Cn + c]);
    resf[(size_t)r * Cn + c] = m;
  }
  if (tid < 3) {
    float s = 0.0f;
    for (int a = 0; a < apr; ++a) s += pos[(size_t)(r * apr + a) * 3 + tid];
    aapos[r * 3 + tid] = s / (float)apr;
  }
  if (tid == 3) {
    int m = 0;
    for (int a = 0; a < apr; ++a) m = max(m, mask[r * apr + a]);
    maskr[r] = m;
  }
}

// ---------------------------------------------------------------------------
// head2: [R,256] @ [256,1] + b, then mask.
// ---------------------------------------------------------------------------
__global__ void head2_kernel(const float* __restrict__ h1, const float* __restrict__ W,
                             const float* __restrict__ b, const int* __restrict__ maskr,
                             float* __restrict__ out, int R, int K) {
  int r = blockIdx.x * blockDim.x + threadIdx.x;
  if (r >= R) return;
  float acc = b[0];
  for (int k = 0; k < K; ++k) acc += h1[(size_t)r * K + k] * W[k];
  out[r] = (maskr[r] == 1) ? acc : 0.0f;
}

// ---------------------------------------------------------------------------
// Host orchestration.
// Input order (jax tree: dicts flatten in sorted-key order):
//  0 x0, 1 pos, 2 normal, 3 aa_norm, 4 pool_batch, 5 aa_num, 6 mask,
//  7 edge1, 8 edge2, 9 edge3, 10 edge4, 11.. params leaves:
//  conv{1..4}: attn.W, attn.b, attn.be, attn.g, attn.rm, attn.rv,
//              lin, lin_dst, lin_src, pos.W, pos.b, pos.be, pos.g, pos.rm, pos.rv  (15 each)
//  head1: W,b,be,g,rm,rv   head2: W,b   neck: W,b,be,g,rm,rv
// ---------------------------------------------------------------------------
extern "C" void kernel_launch(void* const* d_in, const int* in_sizes, int n_in,
                              void* d_out, int out_size, void* d_ws, size_t ws_size,
                              hipStream_t stream) {
  const float* x0    = (const float*)d_in[0];
  const float* pos   = (const float*)d_in[1];
  const float* nrm   = (const float*)d_in[2];
  const float* aanrm = (const float*)d_in[3];
  const int*   mask  = (const int*)d_in[6];

  const int N   = in_sizes[1] / 3;
  const int FIN = in_sizes[0] / N;
  const int R   = in_sizes[3] / 3;
  const int APR = N / R;
  const int KpadIn = (FIN + 31) & ~31;

  int E[4];
  const int *esrc[4], *edst[4];
  for (int i = 0; i < 4; ++i) {
    E[i] = in_sizes[7 + i] / 2;
    esrc[i] = (const int*)d_in[7 + i];
    edst[i] = esrc[i] + E[i];
  }

  const int PB = 11;
  auto P = [&](int i) { return (const float*)d_in[PB + i]; };
  // leaf bases
  const int CB[4]   = {0, 15, 30, 45};  // conv1..conv4
  const int H1B     = 60, H2B = 66, NKB = 68;

  // workspace bump allocator
  char* wsp = (char*)d_ws;
  size_t off = 0;
  auto alloc = [&](size_t bytes) -> void* {
    void* p = wsp + off;
    off = (off + bytes + 255) & ~(size_t)255;
    return p;
  };
  auto packW = [&](const float* W, int K, int Kpad, int Nn) -> _Float16* {
    _Float16* p = (_Float16*)alloc((size_t)Kpad * Nn * sizeof(_Float16));
    int tot = Kpad * Nn;
    pack_b_kernel<<<(tot + 255) / 256, 256, 0, stream>>>(W, p, K, Kpad, Nn);
    return p;
  };
  auto fuseBN = [&](int wleaf, int d) -> float* {
    float* ss = (float*)alloc((size_t)2 * d * sizeof(float));
    fuse_bn_kernel<<<(d + 255) / 256, 256, 0, stream>>>(
        P(wleaf + 1), P(wleaf + 2), P(wleaf + 3), P(wleaf + 4), P(wleaf + 5), ss, d);
    return ss;
  };

  // ---- pack weights & fuse BN ----
  _Float16 *linp[4], *srcp[4], *dstp[4], *attnp[4];
  float *posSS[4], *attnSS[4];
  for (int c = 0; c < 3; ++c) {
    int cb = CB[c];
    attnp[c] = packW(P(cb + 0), 128, 128, 128);
    linp[c]  = packW(P(cb + 6), FIN, KpadIn, 128);
    dstp[c]  = packW(P(cb + 7), FIN, KpadIn, 128);
    srcp[c]  = packW(P(cb + 8), FIN, KpadIn, 128);
    attnSS[c] = fuseBN(cb + 0, 128);
    posSS[c]  = fuseBN(cb + 9, 128);
  }
  {
    int cb = CB[3];
    attnp[3] = packW(P(cb + 0), 512, 512, 512);
    linp[3]  = packW(P(cb + 6), 512, 512, 512);
    dstp[3]  = packW(P(cb + 7), 512, 512, 512);
    srcp[3]  = packW(P(cb + 8), 512, 512, 512);
    attnSS[3] = fuseBN(cb + 0, 512);
    posSS[3]  = fuseBN(cb + 9, 512);
  }
  _Float16* neckp  = packW(P(NKB), 384, 384, 512);
  float*    neckSS = fuseBN(NKB, 512);
  _Float16* h1p    = packW(P(H1B), 512, 512, 256);
  float*    h1SS   = fuseBN(H1B, 256);

  // ---- CSR edge ranges ----
  int *rsA[4], *reA[4];
  for (int i = 0; i < 4; ++i) {
    int n = (i < 3) ? N : R;
    rsA[i] = (int*)alloc((size_t)n * sizeof(int));
    reA[i] = (int*)alloc((size_t)n * sizeof(int));
    edge_range_init<<<(n + 255) / 256, 256, 0, stream>>>(rsA[i], reA[i], n);
    edge_range_build<<<(E[i] + 255) / 256, 256, 0, stream>>>(edst[i], E[i], rsA[i], reA[i]);
  }

  // ---- feature buffers ----
  float* xvb  = (float*)alloc((size_t)N * 128 * sizeof(float));
  float* asb  = (float*)alloc((size_t)N * 128 * sizeof(float));
  float* adb  = (float*)alloc((size_t)N * 128 * sizeof(float));
  float* xcat = (float*)alloc((size_t)N * 384 * sizeof(float));
  float* nout = (float*)alloc((size_t)N * 512 * sizeof(float));
  float* resf = (float*)alloc((size_t)R * 512 * sizeof(float));
  float* aapos= (float*)alloc((size_t)R * 3 * sizeof(float));
  int*   maskr= (int*)alloc((size_t)R * sizeof(int));
  float* xv4  = (float*)alloc((size_t)R * 512 * sizeof(float));
  float* as4  = (float*)alloc((size_t)R * 512 * sizeof(float));
  float* ad4  = (float*)alloc((size_t)R * 512 * sizeof(float));
  float* c4o  = (float*)alloc((size_t)R * 512 * sizeof(float));
  float* h1o  = (float*)alloc((size_t)R * 256 * sizeof(float));

  auto gemm = [&](const float* A, int M, int K, int Kpad, const _Float16* Bp, int Nn,
                  const float* ss, float* Co, int ldo, int col0) {
    dim3 grid(M / 64, Nn / 64);
    wmma_gemm_kernel<<<grid, 128, (size_t)64 * Kpad * sizeof(_Float16), stream>>>(
        A, K, Kpad, Bp, Nn, ss, Co, ldo, col0);
  };

  // ---- atom-level convs 1..3 (buffers reused sequentially) ----
  for (int c = 0; c < 3; ++c) {
    int cb = CB[c];
    gemm(x0, N, FIN, KpadIn, linp[c], 128, nullptr, xvb, 128, 0);
    gemm(x0, N, FIN, KpadIn, srcp[c], 128, nullptr, asb, 128, 0);
    gemm(x0, N, FIN, KpadIn, dstp[c], 128, nullptr, adb, 128, 0);
    edge_conv_kernel<128><<<N, 128, 0, stream>>>(
        asb, adb, xvb, pos, nrm, esrc[c], rsA[c], reA[c],
        P(cb + 9), posSS[c], attnp[c], attnSS[c], xcat, 384, c * 128);
  }

  // ---- neck MLP: [N,384] -> [N,512] with BN+ReLU ----
  gemm(xcat, N, 384, 384, neckp, 512, neckSS, nout, 512, 0);

  // ---- residue pooling ----
  pool_kernel<<<R, 128, 0, stream>>>(nout, pos, mask, APR, 512, resf, aapos, maskr);

  // ---- residue-level conv4 ----
  {
    int cb = CB[3];
    gemm(resf, R, 512, 512, linp[3], 512, nullptr, xv4, 512, 0);
    gemm(resf, R, 512, 512, srcp[3], 512, nullptr, as4, 512, 0);
    gemm(resf, R, 512, 512, dstp[3], 512, nullptr, ad4, 512, 0);
    edge_conv_kernel<512><<<R, 128, 0, stream>>>(
        as4, ad4, xv4, aapos, aanrm, esrc[3], rsA[3], reA[3],
        P(cb + 9), posSS[3], attnp[3], attnSS[3], c4o, 512, 0);
  }

  // ---- heads ----
  gemm(c4o, R, 512, 512, h1p, 256, h1SS, h1o, 256, 0);
  head2_kernel<<<(R + 255) / 256, 256, 0, stream>>>(
      h1o, P(H2B), P(H2B + 1), maskr, (float*)d_out, R, 256);

  (void)n_in; (void)out_size; (void)ws_size; (void)E;
}
